// Net_39041252721058
// MI455X (gfx1250) — compile-verified, matrix operand order checked by reading, not run on hardware
//
#include <hip/hip_runtime.h>
#include <hip/hip_bf16.h>
#include <stdint.h>

// ---------------------------------------------------------------------------
// GCN (2x GraphConvolution) for MI455X / gfx1250.
//   per graph:  H = relu(spmm(A, X@W1) + b1);  out = spmm(A, H@W2) + b2
// Dense GEMMs: V_WMMA_F32_16X16X32_BF16 (wave32 WMMA, f32 accumulate),
//   bf16 operands, all L2-resident (192 MB L2 >> 102 MB working set).
// SpMM: edge-parallel, global_load_b128 row gathers + forced
//   global_atomic_add_f32 scatters (inline asm, non-returning, STOREcnt).
// Elementwise passes: b128-vectorized (bandwidth-bound at 23.3 TB/s).
// ---------------------------------------------------------------------------

typedef __bf16 bf16_t;
typedef __attribute__((ext_vector_type(16))) __bf16 v16bf;
typedef __attribute__((ext_vector_type(8)))  __bf16 v8bf;
typedef __attribute__((ext_vector_type(4)))  __bf16 v4bf;
typedef __attribute__((ext_vector_type(8)))  float  v8f;

#define N_NODES 50000
#define N_EDGES 1600000
#define F_IN    512
#define NHID1   256
#define NHID2   128

// Non-returning f32 atomic add: guaranteed single VMEM instruction.
__device__ __forceinline__ void atom_add_f32(float* p, float v) {
  asm volatile("global_atomic_add_f32 %0, %1, off"
               :
               : "v"((uint64_t)(uintptr_t)p), "v"(v)
               : "memory");
}

// ---------------------------------------------------------------------------
// elementwise helpers (vectorized; all sizes here are multiples of 8)
// ---------------------------------------------------------------------------
__global__ void zero_f32_v4(float4* __restrict__ p, long n4) {
  long i = (long)blockIdx.x * blockDim.x + threadIdx.x;
  if (i < n4) p[i] = make_float4(0.f, 0.f, 0.f, 0.f);
}

// 8 f32 -> 8 bf16 per thread (two b128 loads, one b128 store)
__global__ void cvt_f32_bf16_v8(const float4* __restrict__ x, v8bf* __restrict__ y,
                                long n8) {
  long i = (long)blockIdx.x * blockDim.x + threadIdx.x;
  if (i >= n8) return;
  float4 a = x[2 * i];
  float4 b = x[2 * i + 1];
  v8bf o;
  o[0] = (bf16_t)a.x; o[1] = (bf16_t)a.y; o[2] = (bf16_t)a.z; o[3] = (bf16_t)a.w;
  o[4] = (bf16_t)b.x; o[5] = (bf16_t)b.y; o[6] = (bf16_t)b.z; o[7] = (bf16_t)b.w;
  y[i] = o;
}

// WT[c][r] = (bf16) W[r][c];  W is rows x cols row-major (rows = K, cols = N)
__global__ void transpose_cvt_bf16(const float* __restrict__ W, bf16_t* __restrict__ WT,
                                   int rows, int cols) {
  long i = (long)blockIdx.x * blockDim.x + threadIdx.x;
  if (i >= (long)rows * cols) return;
  int r = (int)(i / cols);
  int c = (int)(i % cols);
  WT[(long)c * rows + r] = (bf16_t)W[(long)r * cols + c];
}

// H[i*4 .. i*4+3] = bf16(relu(T + b));  D4 = D/4 (power of two)
__global__ void relu_bias_cvt_v4(const float4* __restrict__ T4,
                                 const float4* __restrict__ b4,
                                 v4bf* __restrict__ H4, long n4, int D4) {
  long i = (long)blockIdx.x * blockDim.x + threadIdx.x;
  if (i >= n4) return;
  float4 t = T4[i];
  float4 b = b4[(int)(i & (D4 - 1))];
  float x0 = t.x + b.x, x1 = t.y + b.y, x2 = t.z + b.z, x3 = t.w + b.w;
  v4bf o;
  o[0] = (bf16_t)(x0 > 0.f ? x0 : 0.f);
  o[1] = (bf16_t)(x1 > 0.f ? x1 : 0.f);
  o[2] = (bf16_t)(x2 > 0.f ? x2 : 0.f);
  o[3] = (bf16_t)(x3 > 0.f ? x3 : 0.f);
  H4[i] = o;
}

// O[i*4 .. i*4+3] = b[c]   (D4 = D/4, power of two)
__global__ void bias_init_v4(float4* __restrict__ O4, const float4* __restrict__ b4,
                             long n4, int D4) {
  long i = (long)blockIdx.x * blockDim.x + threadIdx.x;
  if (i < n4) O4[i] = b4[(int)(i & (D4 - 1))];
}

// ---------------------------------------------------------------------------
// WMMA GEMM:  C[M,N] (f32) = A[M,K] (bf16 row-major) * B[K,N],
// B supplied transposed: BT[N,K] bf16 row-major (B fragments load with the
// same per-lane striding as A fragments per ISA 7.12.2 wave32 layouts).
// One wave computes a 16 x 64 strip (4 accumulators sharing one A fragment).
// Requires M%16==0, N%64==0, K%32==0 (M=50000, N in {256,128}, K in {512,256}).
// ---------------------------------------------------------------------------
__global__ void __launch_bounds__(256)
gemm_wmma_bf16(const bf16_t* __restrict__ A, const bf16_t* __restrict__ BT,
               float* __restrict__ C, int M, int N, int K) {
  const int lane   = threadIdx.x & 31;
  const int wid    = blockIdx.x * (blockDim.x >> 5) + (threadIdx.x >> 5);
  const int tilesN = N >> 6;                 // 64-wide strips
  const int tilesM = M >> 4;
  if (wid >= tilesM * tilesN) return;
  const int tileM = wid / tilesN;
  const int tileN = wid % tilesN;

  const int grp = lane >> 4;                 // lane group: selects K sub-range
  const int ml  = lane & 15;                 // A: row within tile; B/D: column

  const bf16_t* arow = A + (long)(tileM * 16 + ml) * K;

  v8f acc0 = {0,0,0,0,0,0,0,0};
  v8f acc1 = {0,0,0,0,0,0,0,0};
  v8f acc2 = {0,0,0,0,0,0,0,0};
  v8f acc3 = {0,0,0,0,0,0,0,0};

  for (int kk = 0; kk < K; kk += 32) {
    // A fragment (16x32 bf16): lane grp0 holds K = kk+[0..7] and kk+16+[0..7],
    // grp1 holds K = kk+8+[0..7] and kk+24+[0..7]  (ISA 7.12.2).
    const int kbase = kk + grp * 8;
    v8bf alo = *(const v8bf*)(arow + kbase);
    v8bf ahi = *(const v8bf*)(arow + kbase + 16);
    v16bf afrag = __builtin_shufflevector(alo, ahi,
        0,1,2,3,4,5,6,7, 8,9,10,11,12,13,14,15);

#pragma unroll
    for (int t = 0; t < 4; ++t) {
      const int col = (tileN * 4 + t) * 16 + ml;
      const bf16_t* brow = BT + (long)col * K;
      v8bf blo = *(const v8bf*)(brow + kbase);
      v8bf bhi = *(const v8bf*)(brow + kbase + 16);
      v16bf bfrag = __builtin_shufflevector(blo, bhi,
          0,1,2,3,4,5,6,7, 8,9,10,11,12,13,14,15);
      v8f& acc = (t == 0) ? acc0 : (t == 1) ? acc1 : (t == 2) ? acc2 : acc3;
      acc = __builtin_amdgcn_wmma_f32_16x16x32_bf16(
          /*neg_a=*/false, afrag, /*neg_b=*/false, bfrag,
          /*c_mod=*/(short)0, acc, /*reuse_a=*/false, /*reuse_b=*/false);
    }
  }

  // D layout (f32 16x16): lane = column ml (+grp half), VGPR r = row (grp*8+r).
#pragma unroll
  for (int t = 0; t < 4; ++t) {
    const v8f& acc = (t == 0) ? acc0 : (t == 1) ? acc1 : (t == 2) ? acc2 : acc3;
    const int col = (tileN * 4 + t) * 16 + ml;
#pragma unroll
    for (int r = 0; r < 8; ++r) {
      const int m = tileM * 16 + grp * 8 + r;
      C[(long)m * N + col] = acc[r];
    }
  }
}

// ---------------------------------------------------------------------------
// SpMM: T[dst[e], :] += w[e] * S[src[e], :]   (one wave per edge)
// D is a multiple of 128: each lane handles float4 chunks, fully coalesced.
// ---------------------------------------------------------------------------
__global__ void __launch_bounds__(256)
spmm_atomic(const float* __restrict__ S, const int* __restrict__ src,
            const int* __restrict__ dst, const float* __restrict__ w,
            float* __restrict__ T, int nE, int D) {
  const int lane   = threadIdx.x & 31;
  const int wid    = blockIdx.x * (blockDim.x >> 5) + (threadIdx.x >> 5);
  const int nwaves = gridDim.x * (blockDim.x >> 5);
  const int d4     = D >> 2;
  for (int e = wid; e < nE; e += nwaves) {
    const int   s  = src[e];
    const int   d  = dst[e];
    const float wv = w[e];
    const float4* srow = (const float4*)(S + (long)s * D);
    float*        trow = T + (long)d * D;
    for (int c4 = lane; c4 < d4; c4 += 32) {
      float4 v = srow[c4];                   // global_load_b128 gather
      float* tp = trow + c4 * 4;
      atom_add_f32(tp + 0, wv * v.x);
      atom_add_f32(tp + 1, wv * v.y);
      atom_add_f32(tp + 2, wv * v.z);
      atom_add_f32(tp + 3, wv * v.w);
    }
  }
}

// ---------------------------------------------------------------------------
// driver
// ---------------------------------------------------------------------------
static inline unsigned blocks_for(long n, int thr) {
  return (unsigned)((n + thr - 1) / thr);
}

extern "C" void kernel_launch(void* const* d_in, const int* in_sizes, int n_in,
                              void* d_out, int out_size, void* d_ws, size_t ws_size,
                              hipStream_t stream) {
  (void)in_sizes; (void)n_in; (void)out_size; (void)ws_size;
  const int N = N_NODES, E = N_EDGES;

  // workspace layout (reused by both graphs): ~205 MB
  char* ws = (char*)d_ws;
  size_t off = 0;
  auto take = [&](size_t bytes) {
    void* p = ws + off;
    off += (bytes + 255) & ~(size_t)255;
    return p;
  };
  bf16_t* Xbf = (bf16_t*)take((size_t)N * F_IN  * sizeof(bf16_t));
  bf16_t* W1T = (bf16_t*)take((size_t)F_IN * NHID1 * sizeof(bf16_t));
  bf16_t* W2T = (bf16_t*)take((size_t)NHID1 * NHID2 * sizeof(bf16_t));
  float*  S1  = (float*) take((size_t)N * NHID1 * sizeof(float));
  float*  T1  = (float*) take((size_t)N * NHID1 * sizeof(float));
  bf16_t* Hbf = (bf16_t*)take((size_t)N * NHID1 * sizeof(bf16_t));
  float*  S2  = (float*) take((size_t)N * NHID2 * sizeof(float));

  const int THR = 256;

  for (int g = 0; g < 2; ++g) {
    const float* X   = (const float*)d_in[g == 0 ? 0 : 1];
    const int*   src = (const int*)  d_in[g == 0 ? 2 : 5];
    const int*   dst = (const int*)  d_in[g == 0 ? 3 : 6];
    const float* w   = (const float*)d_in[g == 0 ? 4 : 7];
    const float* W1  = (const float*)d_in[g == 0 ? 8  : 12];
    const float* b1  = (const float*)d_in[g == 0 ? 9  : 13];
    const float* W2  = (const float*)d_in[g == 0 ? 10 : 14];
    const float* b2  = (const float*)d_in[g == 0 ? 11 : 15];
    float* outg = (float*)d_out + (size_t)g * N * NHID2;

    // X -> bf16 (8 elems/thread, b128 traffic)
    cvt_f32_bf16_v8<<<blocks_for((long)N * F_IN / 8, THR), THR, 0, stream>>>(
        (const float4*)X, (v8bf*)Xbf, (long)N * F_IN / 8);
    // W1[K=512, N=256] -> W1T[N][K] bf16
    transpose_cvt_bf16<<<blocks_for((long)F_IN * NHID1, THR), THR, 0, stream>>>(
        W1, W1T, F_IN, NHID1);
    // S1 = Xbf @ W1   [N, 256]
    {
      long waves = (long)(N / 16) * (NHID1 / 64);
      gemm_wmma_bf16<<<blocks_for(waves * 32, THR), THR, 0, stream>>>(
          Xbf, W1T, S1, N, NHID1, F_IN);
    }
    // T1 = 0; T1 += spmm
    zero_f32_v4<<<blocks_for((long)N * NHID1 / 4, THR), THR, 0, stream>>>(
        (float4*)T1, (long)N * NHID1 / 4);
    spmm_atomic<<<blocks_for((long)E * 32, THR), THR, 0, stream>>>(
        S1, src, dst, w, T1, E, NHID1);
    // Hbf = bf16(relu(T1 + b1))
    relu_bias_cvt_v4<<<blocks_for((long)N * NHID1 / 4, THR), THR, 0, stream>>>(
        (const float4*)T1, (const float4*)b1, (v4bf*)Hbf,
        (long)N * NHID1 / 4, NHID1 / 4);
    // W2[K=256, N=128] -> W2T
    transpose_cvt_bf16<<<blocks_for((long)NHID1 * NHID2, THR), THR, 0, stream>>>(
        W2, W2T, NHID1, NHID2);
    // S2 = Hbf @ W2   [N, 128]
    {
      long waves = (long)(N / 16) * (NHID2 / 64);
      gemm_wmma_bf16<<<blocks_for(waves * 32, THR), THR, 0, stream>>>(
          Hbf, W2T, S2, N, NHID2, NHID1);
    }
    // out = b2; out += spmm
    bias_init_v4<<<blocks_for((long)N * NHID2 / 4, THR), THR, 0, stream>>>(
        (float4*)outg, (const float4*)b2, (long)N * NHID2 / 4, NHID2 / 4);
    spmm_atomic<<<blocks_for((long)E * 32, THR), THR, 0, stream>>>(
        S2, src, dst, w, outg, E, NHID2);
  }
}
